// segment_backbone_modified_6889127543363
// MI455X (gfx1250) — matrix operand order repaired and checked by reading, not compiled, not run on hardware
//
#include <hip/hip_runtime.h>
#include <hip/hip_bf16.h>

// CDNA5 / gfx1250: wave32, WMMA 16x16x32 f16 -> f32 accumulate.
typedef __attribute__((ext_vector_type(16))) _Float16 v16h;
typedef __attribute__((ext_vector_type(8)))  float    v8f;

#define KVOL 27

__device__ __forceinline__ int klocal_of(int v, int half) {
  // 16-bit A/B fragment K pattern: V0..V3 -> K 0,2,4,6 (+1), V4..V7 -> K 16,18,20,22 (+1),
  // lane half adds 8.
  return ((v < 4) ? (2 * v) : (8 + 2 * v)) + half * 8;
}

// ---------------------------------------------------------------------------
// Weight packing: f32 W laid out [kk = k*Cin + c][Cout] -> f16 fragment-major
// layout [chunk][ctile][lane][j], j in 0..15.
//   B (KxN=32x16) lane layout: lanes 0-15 hold K=0..7,16..23, lanes 16-31 K=8..15,24..31.
//   kk beyond Ktot zero-padded.
// ---------------------------------------------------------------------------
__global__ void pack_b_kernel(const float* __restrict__ W, _Float16* __restrict__ dst,
                              int Ktot, int Cout, int total) {
  int idx = blockIdx.x * blockDim.x + threadIdx.x;
  if (idx >= total) return;
  int T    = Cout >> 4;
  int j    = idx & 15;
  int lane = (idx >> 4) & 31;
  int ct   = (idx >> 9) % T;
  int chnk = idx / (512 * T);
  int kk   = chnk * 32 + ((lane >> 4) << 3) + (j & 7) + ((j & 8) << 1);
  int d    = (ct << 4) + (lane & 15);
  float v  = (kk < Ktot) ? W[(size_t)kk * Cout + d] : 0.0f;
  dst[idx] = (_Float16)v;
}

// ---------------------------------------------------------------------------
// Gathered submanifold conv + ReLU as tiled WMMA GEMM.
// out[n,:] = relu( sum_{k,c} mask[k,n]*X[nbr[k,n],c] * W[k,c,:] )
// One wave computes a 16-row x COUT tile; reduction dim kk = k*CIN + c walked
// in 32-deep WMMA chunks. Branchless masked gather (mask as 0/1 multiplier —
// loads are always in-range).
// ---------------------------------------------------------------------------
template <int CIN, int COUT, int WAVES>
__global__ __launch_bounds__(WAVES * 32)
void subm_conv_wmma(const float* __restrict__ X,
                    const int* __restrict__ nbr,
                    const unsigned char* __restrict__ mask,  // jax bool = 1 byte
                    const _Float16* __restrict__ Bpack,
                    float* __restrict__ Y, int N) {
  constexpr int KTOT = KVOL * CIN;
  constexpr int KCH  = (KTOT + 31) / 32;   // padded chunk count (CIN=16 -> 14)
  constexpr int T    = COUT / 16;

  const int lane = threadIdx.x & 31;
  const int wave = threadIdx.x >> 5;
  const int rowBase = (blockIdx.x * WAVES + wave) << 4;
  if (rowBase >= N) return;                // uniform per wave: EXEC stays all-ones
  int myrow = rowBase + (lane & 15);
  if (myrow >= N) myrow = N - 1;           // clamp tail rows for gathers
  const int half = lane >> 4;

  v8f acc[T] = {};

  if constexpr (CIN >= 32) {
    // tap-major: one nbr/mask load per tap, CIN/32 WMMA chunks per tap
    for (int k = 0; k < KVOL; ++k) {
      const int base  = k * N + myrow;
      const int nb    = nbr[base];
      const float m   = (float)mask[base];
      const float* xrow = X + (size_t)nb * CIN;
      const _Float16* bk = Bpack + (size_t)k * (CIN / 32) * T * 512;
      if (k + 1 < KVOL)                    // stream next tap's packed B
        __builtin_prefetch(Bpack + (size_t)(k + 1) * (CIN / 32) * T * 512, 0, 0);
#pragma unroll
      for (int cc = 0; cc < CIN / 32; ++cc) {
        v16h a;
#pragma unroll
        for (int v = 0; v < 8; ++v) {
          const int c = cc * 32 + klocal_of(v, half);
          float2 xv = *(const float2*)(xrow + c);
          a[2 * v]     = (_Float16)(xv.x * m);
          a[2 * v + 1] = (_Float16)(xv.y * m);
        }
        const _Float16* bp = bk + (size_t)cc * T * 512 + lane * 16;
#pragma unroll
        for (int t = 0; t < T; ++t) {
          v16h b = *(const v16h*)(bp + t * 512);
          acc[t] = __builtin_amdgcn_wmma_f32_16x16x32_f16(
              false, a, false, b, (short)0, acc[t], false, false);
        }
      }
    }
  } else {
    // CIN==16: padded kk-major chunks (two taps per chunk), branchless tail
    for (int chunk = 0; chunk < KCH; ++chunk) {
      v16h a;
#pragma unroll
      for (int v = 0; v < 8; ++v) {
        const int kk = chunk * 32 + klocal_of(v, half);
        int k = kk / CIN;
        k = k < (KVOL - 1) ? k : (KVOL - 1);           // clamp padded taps
        const int c = kk % CIN;
        const int base = k * N + myrow;
        const int nb   = nbr[base];
        const float m  = (kk < KTOT) ? (float)mask[base] : 0.0f;
        float2 xv = *(const float2*)(X + (size_t)nb * CIN + c);
        a[2 * v]     = (_Float16)(xv.x * m);
        a[2 * v + 1] = (_Float16)(xv.y * m);
      }
      const _Float16* bp = Bpack + (size_t)chunk * T * 512 + lane * 16;
#pragma unroll
      for (int t = 0; t < T; ++t) {
        v16h b = *(const v16h*)(bp + t * 512);
        acc[t] = __builtin_amdgcn_wmma_f32_16x16x32_f16(
            false, a, false, b, (short)0, acc[t], false, false);
      }
    }
  }

  // ---- epilogue: C layout VGPR r -> row (rowBase + half*8 + r), col lane&15 ----
#pragma unroll
  for (int t = 0; t < T; ++t) {
    const int col = (t << 4) + (lane & 15);
#pragma unroll
    for (int r = 0; r < 8; ++r) {
      const int row = rowBase + half * 8 + r;
      if (row < N) {
        float v = acc[t][r];
        v = v > 0.0f ? v : 0.0f;           // fused ReLU
        Y[(size_t)row * COUT + col] = v;
      }
    }
  }
}

// ---------------------------------------------------------------------------
// Decoder merge:  cat = [bottom | lateral]  (N x 2C)
//   out = relu(cat @ Wm)  +  pairsum(cat)   (pairsum_j = cat[2j] + cat[2j+1])
// ---------------------------------------------------------------------------
template <int C, int WAVES>
__global__ __launch_bounds__(WAVES * 32)
void merge_wmma(const float* __restrict__ Xb,
                const float* __restrict__ Xl,
                const _Float16* __restrict__ Bpack,
                float* __restrict__ Y, int N) {
  constexpr int KTOT = 2 * C;
  constexpr int KCH  = KTOT / 32;
  constexpr int T    = C / 16;

  const int lane = threadIdx.x & 31;
  const int wave = threadIdx.x >> 5;
  const int rowBase = (blockIdx.x * WAVES + wave) << 4;
  if (rowBase >= N) return;
  int myrow = rowBase + (lane & 15);
  if (myrow >= N) myrow = N - 1;
  const int half = lane >> 4;

  v8f acc[T] = {};

  for (int chunk = 0; chunk < KCH; ++chunk) {
    v16h a;
#pragma unroll
    for (int v = 0; v < 8; ++v) {
      const int kk = chunk * 32 + klocal_of(v, half);  // even; pair never splits buffers
      const float* src = (kk < C) ? (Xb + (size_t)myrow * C + kk)
                                  : (Xl + (size_t)myrow * C + (kk - C));
      float2 xv = *(const float2*)src;
      a[2 * v]     = (_Float16)xv.x;
      a[2 * v + 1] = (_Float16)xv.y;
    }
    const _Float16* bp = Bpack + (size_t)chunk * T * 512 + lane * 16;
#pragma unroll
    for (int t = 0; t < T; ++t) {
      v16h b = *(const v16h*)(bp + t * 512);
      acc[t] = __builtin_amdgcn_wmma_f32_16x16x32_f16(
          false, a, false, b, (short)0, acc[t], false, false);
    }
  }

#pragma unroll
  for (int t = 0; t < T; ++t) {
    const int j = (t << 4) + (lane & 15);  // output channel
#pragma unroll
    for (int r = 0; r < 8; ++r) {
      const int row = rowBase + half * 8 + r;
      if (row < N) {
        float m = acc[t][r];
        m = m > 0.0f ? m : 0.0f;           // relu only on merge GEMM part
        const int c0 = 2 * j;
        float2 rp = (c0 < C) ? *(const float2*)(Xb + (size_t)row * C + c0)
                             : *(const float2*)(Xl + (size_t)row * C + (c0 - C));
        Y[(size_t)row * C + j] = m + rp.x + rp.y;
      }
    }
  }
}

// ---------------------------------------------------------------------------
extern "C" void kernel_launch(void* const* d_in, const int* in_sizes, int n_in,
                              void* d_out, int out_size, void* d_ws, size_t ws_size,
                              hipStream_t stream) {
  const float*         feat      = (const float*)d_in[0];
  const int*           nbr0      = (const int*)d_in[1];
  const unsigned char* mask0     = (const unsigned char*)d_in[2];
  const int*           nbr1      = (const int*)d_in[3];
  const unsigned char* mask1     = (const unsigned char*)d_in[4];
  const int*           nbr_down  = (const int*)d_in[5];
  const unsigned char* mask_down = (const unsigned char*)d_in[6];
  const int*           nbr_up    = (const int*)d_in[7];
  const unsigned char* mask_up   = (const unsigned char*)d_in[8];
  const float* W_in    = (const float*)d_in[9];
  const float* W_enc1  = (const float*)d_in[10];
  const float* W_down  = (const float*)d_in[11];
  const float* W_enc2  = (const float*)d_in[12];
  const float* W_lat2  = (const float*)d_in[13];
  const float* W_merge2= (const float*)d_in[14];
  const float* W_up2   = (const float*)d_in[15];
  const float* W_lat1  = (const float*)d_in[16];
  const float* W_merge1= (const float*)d_in[17];
  const float* W_up1   = (const float*)d_in[18];

  const int C_IN = 16, C0 = 32, C1 = 64;
  const int N0 = in_sizes[0] / C_IN;       // 100000
  const int N1 = in_sizes[3] / KVOL;       // 20000

  // simple bump allocator over workspace (256B aligned)
  char* ws = (char*)d_ws;
  size_t off = 0;
  auto alloc = [&](size_t bytes) -> void* {
    void* p = ws + off;
    off = (off + bytes + 255) & ~(size_t)255;
    return p;
  };

  auto pack = [&](const float* W, int Ktot, int Cout) -> const _Float16* {
    const int KCH = (Ktot + 31) / 32;
    const int total = KCH * (Cout / 16) * 512;
    _Float16* dst = (_Float16*)alloc((size_t)total * sizeof(_Float16));
    pack_b_kernel<<<(total + 255) / 256, 256, 0, stream>>>(W, dst, Ktot, Cout, total);
    return dst;
  };

  // repack all weights to f16 fragment-major (deterministic, every call)
  const _Float16* pWin   = pack(W_in,    KVOL * C_IN, C0);
  const _Float16* pWenc1 = pack(W_enc1,  KVOL * C0,   C0);
  const _Float16* pWdown = pack(W_down,  KVOL * C0,   C1);
  const _Float16* pWenc2 = pack(W_enc2,  KVOL * C1,   C1);
  const _Float16* pWlat2 = pack(W_lat2,  KVOL * C1,   C1);
  const _Float16* pWm2   = pack(W_merge2, 2 * C1,     C1);
  const _Float16* pWup2  = pack(W_up2,   KVOL * C1,   C0);
  const _Float16* pWlat1 = pack(W_lat1,  KVOL * C0,   C0);
  const _Float16* pWm1   = pack(W_merge1, 2 * C0,     C0);
  const _Float16* pWup1  = pack(W_up1,   KVOL * C0,   C0);

  // activation scratch
  float* x0   = (float*)alloc((size_t)N0 * C0 * 4);
  float* e1   = (float*)alloc((size_t)N0 * C0 * 4);
  float* xd   = (float*)alloc((size_t)N1 * C1 * 4);
  float* e2   = (float*)alloc((size_t)N1 * C1 * 4);
  float* xl2  = (float*)alloc((size_t)N1 * C1 * 4);
  float* xm2  = (float*)alloc((size_t)N1 * C1 * 4);
  float* bot1 = (float*)alloc((size_t)N0 * C0 * 4);
  float* xl1  = (float*)alloc((size_t)N0 * C0 * 4);
  float* xm1  = (float*)alloc((size_t)N0 * C0 * 4);

  constexpr int WAVES = 8;                 // 256 threads = 8 wave32
  const int RPB = WAVES * 16;              // rows per block
  dim3 blk(WAVES * 32);
  dim3 grid0((N0 + RPB - 1) / RPB);
  dim3 grid1((N1 + RPB - 1) / RPB);

  // conv_input + encoder
  subm_conv_wmma<16, 32, WAVES><<<grid0, blk, 0, stream>>>(feat, nbr0, mask0, pWin,   x0, N0);
  subm_conv_wmma<32, 32, WAVES><<<grid0, blk, 0, stream>>>(x0,   nbr0, mask0, pWenc1, e1, N0);
  subm_conv_wmma<32, 64, WAVES><<<grid1, blk, 0, stream>>>(e1,   nbr_down, mask_down, pWdown, xd, N1);
  subm_conv_wmma<64, 64, WAVES><<<grid1, blk, 0, stream>>>(xd,   nbr1, mask1, pWenc2, e2, N1);
  // decoder stage 2
  subm_conv_wmma<64, 64, WAVES><<<grid1, blk, 0, stream>>>(e2,   nbr1, mask1, pWlat2, xl2, N1);
  merge_wmma<64, WAVES><<<grid1, blk, 0, stream>>>(e2, xl2, pWm2, xm2, N1);
  subm_conv_wmma<64, 32, WAVES><<<grid0, blk, 0, stream>>>(xm2,  nbr_up, mask_up, pWup2, bot1, N0);
  // decoder stage 1
  subm_conv_wmma<32, 32, WAVES><<<grid0, blk, 0, stream>>>(e1,   nbr0, mask0, pWlat1, xl1, N0);
  merge_wmma<32, WAVES><<<grid0, blk, 0, stream>>>(bot1, xl1, pWm1, xm1, N0);
  subm_conv_wmma<32, 32, WAVES><<<grid0, blk, 0, stream>>>(xm1,  nbr0, mask0, pWup1, (float*)d_out, N0);
}
